// NLIF_60198261620884
// MI455X (gfx1250) — matrix operand order repaired and checked by reading, not compiled, not run on hardware
//
#include <hip/hip_runtime.h>

// Fused spiking-neuron step for MI455X (gfx1250, wave32, WMMA).
// I_tot = [g | g_fast | I_ext] @ [w*nt ; W_fast ; W_in] + I_tonic, done as one
// GEMM via v_wmma_f32_16x16x32_bf16 with 3-term bf16 hi/lo split (~fp32 accuracy).
// Operands are pre-packed into WMMA fragment-major layout so the GEMM does only
// contiguous 32B/lane global loads (weights stay resident in the 192MB L2).
// Neuron state update is fused into the GEMM epilogue.

typedef __attribute__((ext_vector_type(16))) __bf16 v16bf;
typedef __attribute__((ext_vector_type(8)))  float  v8f;

#define B_DIM 4096
#define N_DIM 2048
#define K3    6144          // 3 * N_DIM (concatenated K)
#define KT    192           // K3 / 32
#define NT    128           // N_DIM / 16
#define BN    (B_DIM * N_DIM)

// ---------------------------------------------------------------------------
// Pack activations [g | g_fast | I_ext] (B x 3K) into bf16 hi/lo, fragment-major.
// A-fragment (16x32 bf16, ISA 7.12.2): lane<16 holds row M=lane, K in {0..7,16..23};
// lane>=16 holds row M=lane-16, K in {8..15,24..31}; 16 bf16 per lane.
// Packed element t = ((mt*KT + ktile)*32 + lane)*16 + j.
// ---------------------------------------------------------------------------
__global__ __launch_bounds__(256) void pack_A(const float* __restrict__ g,
                                              const float* __restrict__ gfast,
                                              const float* __restrict__ iext,
                                              __bf16* __restrict__ Ah,
                                              __bf16* __restrict__ Al,
                                              int total) {
    int t = blockIdx.x * 256 + threadIdx.x;
    if (t >= total) return;
    int j     = t & 15;
    int lane  = (t >> 4) & 31;
    int tile  = t >> 9;
    int ktile = tile % KT;
    int mt    = tile / KT;
    int kk    = (lane < 16) ? ((j < 8) ? j : j + 8)
                            : ((j < 8) ? j + 8 : j + 16);
    int k = ktile * 32 + kk;
    int m = mt * 16 + (lane & 15);
    float x;
    if (k < N_DIM)            x = g    [m * N_DIM + k];
    else if (k < 2 * N_DIM)   x = gfast[m * N_DIM + (k - N_DIM)];
    else                      x = iext [m * N_DIM + (k - 2 * N_DIM)];
    __bf16 h = (__bf16)x;
    float  r = x - (float)h;
    Ah[t] = h;
    Al[t] = (__bf16)r;
}

// ---------------------------------------------------------------------------
// Pack weights [w*nt ; W_fast ; W_in] (3K x N) into bf16 hi/lo, fragment-major.
// B-fragment (32x16 bf16): lane<16 holds col N=lane, K in {0..7,16..23};
// lane>=16 holds col N=lane-16, K in {8..15,24..31} (same K order as A).
// Packed element t = ((ktile*NT + ntile)*32 + lane)*16 + j.
// ---------------------------------------------------------------------------
__global__ __launch_bounds__(256) void pack_W(const float* __restrict__ w,
                                              const float* __restrict__ wfast,
                                              const float* __restrict__ win,
                                              const float* __restrict__ ntypes,
                                              __bf16* __restrict__ Wh,
                                              __bf16* __restrict__ Wl,
                                              int total) {
    int t = blockIdx.x * 256 + threadIdx.x;
    if (t >= total) return;
    int j     = t & 15;
    int lane  = (t >> 4) & 31;
    int tile  = t >> 9;
    int ntile = tile % NT;
    int ktile = tile / NT;
    int k = ktile * 32 + ((j < 8) ? j : j + 8) + ((lane >> 4) << 3);
    int n = ntile * 16 + (lane & 15);
    float x;
    if (k < N_DIM)            x = w    [k * N_DIM + n] * ntypes[k];
    else if (k < 2 * N_DIM)   x = wfast[(k - N_DIM) * N_DIM + n];
    else                      x = win  [(k - 2 * N_DIM) * N_DIM + n];
    __bf16 h = (__bf16)x;
    float  r = x - (float)h;
    Wh[t] = h;
    Wl[t] = (__bf16)r;
}

// ---------------------------------------------------------------------------
// Fused GEMM + neuron update. Block = 256 threads = 8 waves (2M x 4N);
// each wave owns a 32(M) x 64(N) C tile = 2x4 grid of 16x16 WMMA accumulators.
// Grid = (N/256, B/64). K-loop: 192 steps of K=32, 3 bf16 WMMAs per tile/step.
// ---------------------------------------------------------------------------
__global__ __launch_bounds__(256) void gemm_neuron(
    const v16bf* __restrict__ Ah, const v16bf* __restrict__ Al,
    const v16bf* __restrict__ Wh, const v16bf* __restrict__ Wl,
    const float* __restrict__ v_in, const float* __restrict__ s_in,
    const float* __restrict__ th_in, const float* __restrict__ g_in,
    const float* __restrict__ gf_in, const float* __restrict__ I_tonic,
    const float* __restrict__ Ovec, float* __restrict__ out) {

    const int lane  = threadIdx.x & 31;
    const int wave  = threadIdx.x >> 5;          // 0..7
    const int waveM = wave >> 2;                 // 0..1
    const int waveN = wave & 3;                  // 0..3
    const int mt0 = blockIdx.y * 4 + waveM * 2;  // 16-row tile index (2 tiles)
    const int nt0 = blockIdx.x * 16 + waveN * 4; // 16-col tile index (4 tiles)

    v8f acc[2][4];
#pragma unroll
    for (int i = 0; i < 2; ++i)
#pragma unroll
        for (int j = 0; j < 4; ++j) acc[i][j] = (v8f){0.f,0.f,0.f,0.f,0.f,0.f,0.f,0.f};

    for (int kt = 0; kt < KT; ++kt) {
        v16bf ah[2], al[2], bh[4], bl[4];
#pragma unroll
        for (int i = 0; i < 2; ++i) {
            size_t idx = ((size_t)(mt0 + i) * KT + kt) * 32 + lane;
            ah[i] = Ah[idx];
            al[i] = Al[idx];
        }
#pragma unroll
        for (int j = 0; j < 4; ++j) {
            size_t idx = ((size_t)kt * NT + (nt0 + j)) * 32 + lane;
            bh[j] = Wh[idx];
            bl[j] = Wl[idx];
        }
        if (kt + 1 < KT) {
            __builtin_prefetch((const void*)&Ah[((size_t)mt0 * KT + kt + 1) * 32 + lane], 0, 0);
            __builtin_prefetch((const void*)&Wh[((size_t)(kt + 1) * NT + nt0) * 32 + lane], 0, 0);
        }
#pragma unroll
        for (int i = 0; i < 2; ++i) {
#pragma unroll
            for (int j = 0; j < 4; ++j) {
                acc[i][j] = __builtin_amdgcn_wmma_f32_16x16x32_bf16(
                    false, ah[i], false, bh[j], (short)0, acc[i][j], false, false);
                acc[i][j] = __builtin_amdgcn_wmma_f32_16x16x32_bf16(
                    false, ah[i], false, bl[j], (short)0, acc[i][j], false, false);
                acc[i][j] = __builtin_amdgcn_wmma_f32_16x16x32_bf16(
                    false, al[i], false, bh[j], (short)0, acc[i][j], false, false);
            }
        }
    }

    // Epilogue: C layout per ISA 7.12.2 — lane%16 = column, VGPR r -> row r (+8 for lane>=16).
#pragma unroll
    for (int i = 0; i < 2; ++i) {
#pragma unroll
        for (int j = 0; j < 4; ++j) {
            const int bBase = (mt0 + i) * 16 + ((lane >> 4) << 3);
            const int n     = (nt0 + j) * 16 + (lane & 15);
            const float iton = I_tonic[n];
            const float Oval = Ovec[n];
#pragma unroll
            for (int r = 0; r < 8; ++r) {
                const int b = bBase + r;
                const size_t off = (size_t)b * N_DIM + n;
                float Itot = acc[i][j][r] + iton;
                float vv = v_in[off];
                float ss = s_in[off];
                float th = th_in[off];
                float gg = g_in[off];
                float gf = gf_in[off];
                float dv = (Itot - vv) * 0.1f;           // (V_RESET - v + I_tot)/TAU_M
                float vn = vv + dv;
                float gating = fminf(fmaxf(vn, 0.f), 1.f);
                float sn = ss + (-ss + gating * fminf(fmaxf(dv, 0.f), 1.f)) * 0.1f;
                float spk  = (vn >= th) ? 1.f : 0.f;
                float nspk = 1.f - spk;
                float gn   = spk + nspk * (-gg * 0.1f);
                float gfn  = spk + nspk * (-gf);
                float thn  = 0.7f * th + spk * 0.1f;
                float vres = 0.14f * vv - 0.12f;         // E_L + F_V*(v-E_L) - DELTA_V
                float vnew = spk * vres + nspk * vn;
                out[0 * BN + off] = Oval * sn;   // readout
                out[1 * BN + off] = vnew;
                out[2 * BN + off] = gn;
                out[3 * BN + off] = gfn;
                out[4 * BN + off] = sn;
                out[5 * BN + off] = thn;
            }
        }
    }
}

extern "C" void kernel_launch(void* const* d_in, const int* in_sizes, int n_in,
                              void* d_out, int out_size, void* d_ws, size_t ws_size,
                              hipStream_t stream) {
    const float* I_ext   = (const float*)d_in[0];
    const float* v       = (const float*)d_in[1];
    const float* g       = (const float*)d_in[2];
    const float* g_fast  = (const float*)d_in[3];
    const float* s       = (const float*)d_in[4];
    const float* theta_s = (const float*)d_in[5];
    const float* w       = (const float*)d_in[6];
    const float* W_fast  = (const float*)d_in[7];
    const float* W_in    = (const float*)d_in[8];
    const float* I_tonic = (const float*)d_in[9];
    const float* O       = (const float*)d_in[10];
    const float* ntypes  = (const float*)d_in[11];
    float* out = (float*)d_out;

    // Workspace layout: Ah, Al (B x 3K bf16 each), Wh, Wl (3K x N bf16 each) ~151 MB
    char* ws = (char*)d_ws;
    const size_t szA = (size_t)B_DIM * K3 * sizeof(__bf16);   // 50.3 MB
    const size_t szW = (size_t)K3 * N_DIM * sizeof(__bf16);   // 25.2 MB
    __bf16* Ah = (__bf16*)(ws);
    __bf16* Al = (__bf16*)(ws + szA);
    __bf16* Wh = (__bf16*)(ws + 2 * szA);
    __bf16* Wl = (__bf16*)(ws + 2 * szA + szW);

    const int totA = B_DIM * K3;      // 25,165,824 (multiple of 256)
    const int totW = K3 * N_DIM;      // 12,582,912 (multiple of 256)
    pack_A<<<totA / 256, 256, 0, stream>>>(g, g_fast, I_ext, Ah, Al, totA);
    pack_W<<<totW / 256, 256, 0, stream>>>(w, W_fast, W_in, ntypes, Wh, Wl, totW);

    dim3 grid(N_DIM / 256, B_DIM / 64);   // (8, 64)
    gemm_neuron<<<grid, 256, 0, stream>>>((const v16bf*)Ah, (const v16bf*)Al,
                                          (const v16bf*)Wh, (const v16bf*)Wl,
                                          v, s, theta_s, g, g_fast, I_tonic, O, out);
}
